// KANLayer_28097676050639
// MI455X (gfx1250) — matrix operand order repaired and checked by reading, not compiled, not run on hardware
//
#include <hip/hip_runtime.h>
#include <hip/hip_bf16.h>
#include <stdint.h>

typedef __bf16 bf16;
typedef __attribute__((ext_vector_type(16))) __bf16 bf16x16;
typedef __attribute__((ext_vector_type(8)))  __bf16 bf16x8;
typedef __attribute__((ext_vector_type(8)))  float  f32x8;

#define BATCH    8192
#define IN_FEAT  512
#define OUT_FEAT 512
#define NPLANES  14
#define KTOT     (NPLANES * IN_FEAT)   // 7168

#define BM 128
#define BN 128
#define BK 32
#define LDST 40   // padded LDS row stride in bf16 (80 B -> conflict-free b128 reads)

// ---------------------------------------------------------------------------
// Kernel 1: pack base_w [512,512] and spline_w [512,512,13] into bf16
// Wpack[o, p*512 + i], p=0 -> base_w, p=1..13 -> spline_w[..,p-1]
// ---------------------------------------------------------------------------
__global__ void kan_pack_w(const float* __restrict__ base_w,
                           const float* __restrict__ spline_w,
                           bf16* __restrict__ Wpack) {
    int o = blockIdx.y;                         // 0..511
    int r = blockIdx.x * 256 + threadIdx.x;     // 0..7167
    int p = r >> 9;                             // plane 0..13
    int i = r & 511;
    float w = (p == 0) ? base_w[o * IN_FEAT + i]
                       : spline_w[((size_t)o * IN_FEAT + i) * 13 + (p - 1)];
    Wpack[(size_t)o * KTOT + r] = (bf16)w;
}

// ---------------------------------------------------------------------------
// Kernel 2: per (b,i) compute silu(x) and the 13 cubic B-spline bases
// (uniform knots t_j = -3.2 + 0.4 j), write plane-major bf16 rows of A.
// ---------------------------------------------------------------------------
__global__ void kan_prep_a(const float* __restrict__ x,
                           bf16* __restrict__ Apack) {
    int idx = blockIdx.x * 256 + threadIdx.x;   // b*512 + i
    int b = idx >> 9;
    int i = idx & 511;
    float xv = x[idx];

    float s = xv / (1.0f + __expf(-xv));        // silu
    bf16* row = Apack + (size_t)b * KTOT + i;
    row[0] = (bf16)s;

    float basis[16];
#pragma unroll
    for (int j = 0; j < 16; ++j) {
        float t0 = -3.2f + 0.4f * j;
        float t1 = t0 + 0.4f;
        basis[j] = (xv >= t0 && xv < t1) ? 1.0f : 0.0f;
    }
#pragma unroll
    for (int p = 1; p <= 3; ++p) {
        float inv = 1.0f / (0.4f * (float)p + 1e-8f);
#pragma unroll
        for (int j = 0; j < 16 - p; ++j) {
            float tj  = -3.2f + 0.4f * j;
            float tjp = tj + 0.4f * (float)p;
            basis[j] = (xv - tj) * inv * basis[j] +
                       (tjp + 0.4f - xv) * inv * basis[j + 1];
        }
    }
#pragma unroll
    for (int g = 0; g < 13; ++g)
        row[(g + 1) * IN_FEAT] = (bf16)basis[g];
}

// ---------------------------------------------------------------------------
// Kernel 3: bf16 WMMA GEMM, triple-buffered LDS tiles filled by
// GLOBAL_LOAD_ASYNC_TO_LDS_B128 (ASYNCcnt), 2-step lookahead, 1 barrier/step.
// Block tile 128x128x32 with 4 waves (2M x 2N), wave tile 64x64 (4x4 WMMA
// tiles -> 16 wmma per 16 ds_load_b128, 2x better LDS-read density).
// ---------------------------------------------------------------------------
__global__ __launch_bounds__(128)
void kan_gemm(const bf16* __restrict__ A, const bf16* __restrict__ W,
              const float* __restrict__ base_b, float* __restrict__ out) {
    __shared__ __attribute__((aligned(16))) bf16 As0[BM * LDST];
    __shared__ __attribute__((aligned(16))) bf16 As1[BM * LDST];
    __shared__ __attribute__((aligned(16))) bf16 As2[BM * LDST];
    __shared__ __attribute__((aligned(16))) bf16 Bs0[BN * LDST];
    __shared__ __attribute__((aligned(16))) bf16 Bs1[BN * LDST];
    __shared__ __attribute__((aligned(16))) bf16 Bs2[BN * LDST];   // 60 KB

    const int t    = threadIdx.x;        // 0..127
    const int wave = t >> 5;             // 0..3
    const int lane = t & 31;
    const int wm   = wave >> 1;          // 0..1  (M dir)
    const int wn   = wave & 1;           // 0..1  (N dir)
    const int l16  = lane & 15;
    const int half = lane >> 4;          // 0 or 1

    const int bm0 = blockIdx.y * BM;
    const int bn0 = blockIdx.x * BN;

    // cooperative copy: thread t owns full 64-byte k-chunk of row t (A and B)
    uint32_t gA = (uint32_t)(bm0 + t) * (KTOT * 2);
    uint32_t gW = (uint32_t)(bn0 + t) * (KTOT * 2);

    const uint32_t lA0 = (uint32_t)(uintptr_t)&As0[t * LDST];
    const uint32_t lA1 = (uint32_t)(uintptr_t)&As1[t * LDST];
    const uint32_t lA2 = (uint32_t)(uintptr_t)&As2[t * LDST];
    const uint32_t lB0 = (uint32_t)(uintptr_t)&Bs0[t * LDST];
    const uint32_t lB1 = (uint32_t)(uintptr_t)&Bs1[t * LDST];
    const uint32_t lB2 = (uint32_t)(uintptr_t)&Bs2[t * LDST];

    // one tile = 8 async b128 per thread (A row 64B + B row 64B)
    auto issue = [&](uint32_t la, uint32_t lb) {
        asm volatile("global_load_async_to_lds_b128 %0, %1, %2"
                     :: "v"(la), "v"(gA), "s"(A) : "memory");
        asm volatile("global_load_async_to_lds_b128 %0, %1, %2 offset:16"
                     :: "v"(la), "v"(gA), "s"(A) : "memory");
        asm volatile("global_load_async_to_lds_b128 %0, %1, %2 offset:32"
                     :: "v"(la), "v"(gA), "s"(A) : "memory");
        asm volatile("global_load_async_to_lds_b128 %0, %1, %2 offset:48"
                     :: "v"(la), "v"(gA), "s"(A) : "memory");
        asm volatile("global_load_async_to_lds_b128 %0, %1, %2"
                     :: "v"(lb), "v"(gW), "s"(W) : "memory");
        asm volatile("global_load_async_to_lds_b128 %0, %1, %2 offset:16"
                     :: "v"(lb), "v"(gW), "s"(W) : "memory");
        asm volatile("global_load_async_to_lds_b128 %0, %1, %2 offset:32"
                     :: "v"(lb), "v"(gW), "s"(W) : "memory");
        asm volatile("global_load_async_to_lds_b128 %0, %1, %2 offset:48"
                     :: "v"(lb), "v"(gW), "s"(W) : "memory");
        gA += BK * 2;
        gW += BK * 2;
    };

    f32x8 acc[4][4];
#pragma unroll
    for (int i2 = 0; i2 < 4; ++i2)
#pragma unroll
        for (int j = 0; j < 4; ++j)
            acc[i2][j] = (f32x8)(0.0f);

    auto compute = [&](const bf16* __restrict__ as, const bf16* __restrict__ bs) {
        bf16x16 afrag[4];
#pragma unroll
        for (int i2 = 0; i2 < 4; ++i2) {
            const bf16* pA = as + (wm * 64 + i2 * 16 + l16) * LDST + half * 8;
            *((bf16x8*)&afrag[i2])     = *(const bf16x8*)(pA);
            *((bf16x8*)&afrag[i2] + 1) = *(const bf16x8*)(pA + 16);
        }
        bf16x16 bfrag[4];
#pragma unroll
        for (int j = 0; j < 4; ++j) {
            const bf16* pB = bs + (wn * 64 + j * 16 + l16) * LDST + half * 8;
            *((bf16x8*)&bfrag[j])     = *(const bf16x8*)(pB);
            *((bf16x8*)&bfrag[j] + 1) = *(const bf16x8*)(pB + 16);
        }
#pragma unroll
        for (int i2 = 0; i2 < 4; ++i2)
#pragma unroll
            for (int j = 0; j < 4; ++j)
                acc[i2][j] = __builtin_amdgcn_wmma_f32_16x16x32_bf16(
                    false, afrag[i2], false, bfrag[j],
                    (short)0, acc[i2][j], false, false);
    };

    // prologue: tiles 0 and 1 in flight
    issue(lA0, lB0);
    issue(lA1, lB1);

    // iter i: wait(tile i landed), barrier, issue(tile i+2), compute(tile i)
    // 224 steps = 74 * 3 (static rotation) + 2 peeled tail steps
    for (int it = 0; it < 74; ++it) {
        asm volatile("s_wait_asynccnt 0x8" ::: "memory");
        __syncthreads();
        issue(lA2, lB2);
        compute(As0, Bs0);

        asm volatile("s_wait_asynccnt 0x8" ::: "memory");
        __syncthreads();
        issue(lA0, lB0);
        compute(As1, Bs1);

        asm volatile("s_wait_asynccnt 0x8" ::: "memory");
        __syncthreads();
        issue(lA1, lB1);
        compute(As2, Bs2);
    }
    asm volatile("s_wait_asynccnt 0x8" ::: "memory");
    __syncthreads();
    compute(As0, Bs0);

    asm volatile("s_wait_asynccnt 0x0" ::: "memory");
    __syncthreads();
    compute(As1, Bs1);

    // epilogue: C layout — VGPR v: row = v + 8*half, col = l16
#pragma unroll
    for (int i2 = 0; i2 < 4; ++i2) {
        int mrow0 = bm0 + wm * 64 + i2 * 16 + half * 8;
#pragma unroll
        for (int j = 0; j < 4; ++j) {
            int ncol = bn0 + wn * 64 + j * 16 + l16;
            float bb = base_b[ncol];
            float* op = out + (size_t)mrow0 * OUT_FEAT + ncol;
#pragma unroll
            for (int v = 0; v < 8; ++v)
                op[(size_t)v * OUT_FEAT] = acc[i2][j][v] + bb;
        }
    }
}

// ---------------------------------------------------------------------------
extern "C" void kernel_launch(void* const* d_in, const int* in_sizes, int n_in,
                              void* d_out, int out_size, void* d_ws, size_t ws_size,
                              hipStream_t stream) {
    const float* x        = (const float*)d_in[0];
    const float* base_w   = (const float*)d_in[1];
    const float* base_b   = (const float*)d_in[2];
    const float* spline_w = (const float*)d_in[3];
    float* out = (float*)d_out;

    bf16* Wpack = (bf16*)d_ws;                                        // 7.34 MB
    bf16* Apack = (bf16*)((char*)d_ws + (size_t)OUT_FEAT * KTOT * 2); // 117.4 MB

    kan_pack_w<<<dim3(KTOT / 256, OUT_FEAT), 256, 0, stream>>>(base_w, spline_w, Wpack);
    kan_prep_a<<<dim3((BATCH * IN_FEAT) / 256), 256, 0, stream>>>(x, Apack);
    kan_gemm<<<dim3(OUT_FEAT / BN, BATCH / BM), 128, 0, stream>>>(Apack, Wpack, base_b, out);
}
